// FFTResonanceLookup_13202729468282
// MI455X (gfx1250) — matrix operand order repaired and checked by reading, not compiled, not run on hardware
//
#include <hip/hip_runtime.h>
#include <hip/hip_bf16.h>
#include <math.h>

typedef __attribute__((ext_vector_type(8)))  _Float16 v8h;
typedef __attribute__((ext_vector_type(16))) _Float16 v16h;
typedef __attribute__((ext_vector_type(8)))  float    v8f;

#define NROW   1024      // 8*32*4
#define NITEM  4096
#define NCOL   514       // 2*NC
#define NC     257
#define NPAD   528       // NCOL padded to 16
#define NFRM   128
#define KA     544       // [Sr(272) | Si(272)]
#define K2     1088      // [S[f] | S[f-1]]
#define ASTRIDE 552      // LDS row stride (halfs), 16B-aligned rows
#define PI_F   3.14159265358979323846f

__device__ __forceinline__ v16h combine16(v8h a, v8h b) {
    v16h r;
#pragma unroll
    for (int i = 0; i < 8; ++i) { r[i] = a[i]; r[i + 8] = b[i]; }
    return r;
}

// ---------------------------------------------------------------------------
// K0: windowed twiddle basis Bt2[t'][k], t' in [0,256), k in [0,1088)
//   k < 544  : frame f  part, t = t'        (Sr*cos - Si*sin terms)
//   k >= 544 : frame f-1 part, t = t' + 256
//   within a part: kk<272 -> cos column c=kk ; kk>=272 -> -sin column c=kk-272
// ---------------------------------------------------------------------------
__global__ void twiddle_kernel(_Float16* __restrict__ Bt2) {
    const int tp = blockIdx.x;                  // t' 0..255
    for (int k = threadIdx.x; k < K2; k += blockDim.x) {
        const int part = (k >= KA) ? 1 : 0;
        const int t    = tp + part * 256;       // 0..511
        const int kk   = k - part * KA;
        const int isSin = (kk >= 272) ? 1 : 0;
        const int c    = isSin ? (kk - 272) : kk;
        float v = 0.0f;
        if (c <= 256) {
            const float a   = (c == 0 || c == 256) ? (1.0f / 512.0f) : (2.0f / 512.0f);
            const float win = 0.5f - 0.5f * cosf(2.0f * PI_F * (float)t / 512.0f);
            const int   ct  = (c * t) & 511;    // exact range reduction
            const float ang = 2.0f * PI_F * (float)ct / 512.0f;
            v = isSin ? (-a * win * sinf(ang)) : (a * win * cosf(ang));
        }
        Bt2[(size_t)tp * K2 + k] = (_Float16)v;
    }
}

// ---------------------------------------------------------------------------
// K1: row softmax over 4096 items -> f16 weights
// ---------------------------------------------------------------------------
__global__ __launch_bounds__(256) void softmax_kernel(const float* __restrict__ sel,
                                                      _Float16* __restrict__ wH) {
    __shared__ float red[256];
    const int r = blockIdx.x, tid = threadIdx.x;
    const float* row = sel + (size_t)r * NITEM;

    float mx = -3.4e38f;
    for (int i = tid; i < NITEM; i += 256) mx = fmaxf(mx, row[i]);
    red[tid] = mx; __syncthreads();
    for (int s = 128; s > 0; s >>= 1) { if (tid < s) red[tid] = fmaxf(red[tid], red[tid + s]); __syncthreads(); }
    mx = red[0]; __syncthreads();

    float sum = 0.0f;
    for (int i = tid; i < NITEM; i += 256) sum += expf(row[i] - mx);
    red[tid] = sum; __syncthreads();
    for (int s = 128; s > 0; s >>= 1) { if (tid < s) red[tid] += red[tid + s]; __syncthreads(); }
    const float inv = 1.0f / red[0];

    for (int i = tid; i < NITEM; i += 256)
        wH[(size_t)r * NITEM + i] = (_Float16)(expf(row[i] - mx) * inv);
}

// ---------------------------------------------------------------------------
// K2: items (4096 x 514 f32) -> itemsT (528 x 4096 f16), pad cols with 0
// ---------------------------------------------------------------------------
__global__ void transpose_items_kernel(const float* __restrict__ items,
                                       _Float16* __restrict__ itemsT) {
    const int c = blockIdx.x;                   // 0..527
    for (int i = threadIdx.x; i < NITEM; i += blockDim.x) {
        float v = (c < NCOL) ? items[(size_t)i * NCOL + c] : 0.0f;
        itemsT[(size_t)c * NITEM + i] = (_Float16)v;
    }
}

// ---------------------------------------------------------------------------
// K3: selected = w @ items  (M=1024, K=4096, N=528) via WMMA f16->f32
//     each wave: 4 M-tiles x 1 N-tile (B fragment reused 4x)
//     16 M-blocks x 33 N-tiles = 528 waves = 66 WGs
// ---------------------------------------------------------------------------
__global__ __launch_bounds__(256) void sel_gemm_kernel(const _Float16* __restrict__ wH,
                                                       const _Float16* __restrict__ itemsT,
                                                       float* __restrict__ selected) {
    const int tid  = threadIdx.x;
    const int lane = tid & 31;
    const int waveId = blockIdx.x * 8 + (tid >> 5);   // 0..527
    const int mb = waveId & 15;                       // M-block of 64 rows
    const int nt = waveId >> 4;                       // 0..32
    const int n0 = nt * 16;
    const int m = lane & 15, h = lane >> 4;

    const _Float16* arow0 = wH     + (size_t)(mb * 64 + m) * NITEM;  // A rows (M=lane&15)
    const _Float16* brow  = itemsT + (size_t)(n0 + m) * NITEM;       // B col  (N=lane&15)

    v8f acc[4] = {};
    for (int ks = 0; ks < NITEM / 32; ++ks) {
        const int k0 = ks * 32;
        // B layout: lane half h holds contiguous K [16h, 16h+16)
        v8h b0 = *(const v8h*)&brow[k0 + 16 * h];
        v8h b1 = *(const v8h*)&brow[k0 + 16 * h + 8];
        const v16h B = combine16(b0, b1);
#pragma unroll
        for (int q = 0; q < 4; ++q) {
            const _Float16* arow = arow0 + (size_t)q * 16 * NITEM;
            // A layout: lane half h holds contiguous K [16h, 16h+16)
            v8h a0 = *(const v8h*)&arow[k0 + 16 * h];
            v8h a1 = *(const v8h*)&arow[k0 + 16 * h + 8];
            acc[q] = __builtin_amdgcn_wmma_f32_16x16x32_f16(false, combine16(a0, a1),
                                                            false, B,
                                                            (short)0, acc[q], false, false);
        }
    }
    // C layout: VGPR j -> M = j + 8h, N = lane&15
#pragma unroll
    for (int q = 0; q < 4; ++q) {
        float* op = selected + (size_t)(mb * 64 + q * 16 + 8 * h) * NPAD + n0 + m;
#pragma unroll
        for (int j = 0; j < 8; ++j) op[(size_t)j * NPAD] = acc[q][j];
    }
}

// ---------------------------------------------------------------------------
// K4: per-row resonance.  Build S (128 x 544 f16) in LDS with the mag-power
//     recurrence, then out-block GEMM:  out[f*256+t'] = [S[f]|S[f-1]] . Bt2[t']
//     Each wave: 2 N-tiles x all 8 M-tiles (B fragment reused 8x, A from LDS).
//     LDS has a zero guard row at index 0 so S[-1] == 0 without divergence.
// ---------------------------------------------------------------------------
__global__ __launch_bounds__(256) void resonance_kernel(const float* __restrict__ selected,
                                                        const _Float16* __restrict__ Bt2,
                                                        float* __restrict__ out) {
    extern __shared__ _Float16 Alds[];          // (128+1) * ASTRIDE halfs
    const int r = blockIdx.x;
    const int tid = threadIdx.x;

    // zero the guard row (S[-1])
    for (int i = tid; i < ASTRIDE; i += 256) Alds[i] = (_Float16)0.0f;

    // Phase A: generate S rows (frame-major, [Sr(272)|Si(272)] per row)
    for (int c = tid; c < 272; c += 256) {
        if (c <= 256) {
            const float sm  = selected[(size_t)r * NPAD + c];
            const float sp  = selected[(size_t)r * NPAD + NC + c];
            const float mag = 0.9999f / (1.0f + expf(-sm));
            const float ph  = tanhf(sp) * PI_F;
            const float cp  = cosf(ph);
            const float spn = sinf(ph);
            float p = 1.0f;
#pragma unroll 4
            for (int f = 0; f < NFRM; ++f) {
                p *= mag;                                     // mag^(f+1)
                Alds[(size_t)(f + 1) * ASTRIDE + c]       = (_Float16)(p * cp);
                Alds[(size_t)(f + 1) * ASTRIDE + 272 + c] = (_Float16)(p * spn);
            }
        } else {
            for (int f = 0; f < NFRM; ++f) {
                Alds[(size_t)(f + 1) * ASTRIDE + c]       = (_Float16)0.0f;
                Alds[(size_t)(f + 1) * ASTRIDE + 272 + c] = (_Float16)0.0f;
            }
        }
    }
    __syncthreads();

    // Phase B: WMMA GEMM, 2 N-tiles per wave, all 8 M-tiles per B fragment
    const int lane = tid & 31;
    const int wv   = tid >> 5;          // 0..7
    const int m    = lane & 15;
    const int h    = lane >> 4;

    for (int ntp = 0; ntp < 2; ++ntp) {
        const int nt = wv + ntp * 8;                               // N-tile
        const _Float16* brow = Bt2 + (size_t)(nt * 16 + m) * K2;   // B column t'
        v8f acc[8] = {};
        for (int ks = 0; ks < 34; ++ks) {
            const int k0   = ks * 32;
            const int part = (ks >= 17) ? 1 : 0;    // KA=544 is a multiple of 32
            const int kk   = k0 - part * KA;
            v8h b0 = *(const v8h*)&brow[k0 + 16 * h];
            v8h b1 = *(const v8h*)&brow[k0 + 16 * h + 8];
            const v16h B = combine16(b0, b1);
            __builtin_prefetch(&brow[k0 + 64], 0, 1);   // global_prefetch_b8 (next lines)
#pragma unroll
            for (int mt = 0; mt < 8; ++mt) {
                const int fr = mt * 16 + m - part + 1;  // +1 guard row offset
                v8h a0 = *(const v8h*)&Alds[(size_t)fr * ASTRIDE + kk + 16 * h];
                v8h a1 = *(const v8h*)&Alds[(size_t)fr * ASTRIDE + kk + 16 * h + 8];
                acc[mt] = __builtin_amdgcn_wmma_f32_16x16x32_f16(false, combine16(a0, a1),
                                                                 false, B,
                                                                 (short)0, acc[mt], false, false);
            }
        }
        // C tiles -> out[r][(mt*16+M)*256 + nt*16 + N], M = j + 8h, N = m
#pragma unroll
        for (int mt = 0; mt < 8; ++mt) {
            float* op = out + (size_t)r * 32768 + (size_t)(mt * 16 + 8 * h) * 256 + nt * 16 + m;
#pragma unroll
            for (int j = 0; j < 8; ++j) op[(size_t)j * 256] = acc[mt][j];
        }
    }
}

// ---------------------------------------------------------------------------
extern "C" void kernel_launch(void* const* d_in, const int* in_sizes, int n_in,
                              void* d_out, int out_size, void* d_ws, size_t ws_size,
                              hipStream_t stream) {
    const float* selections = (const float*)d_in[0];   // (8,32,4,4096) f32
    const float* items      = (const float*)d_in[1];   // (4096,514)    f32
    float* out = (float*)d_out;                        // 1024 * 32768  f32

    char* ws = (char*)d_ws;
    _Float16* Bt2 = (_Float16*)ws;                                  // 256*1088*2   = 557056
    _Float16* wH  = (_Float16*)(ws + 557056);                       // 1024*4096*2  = 8388608
    _Float16* itT = (_Float16*)(ws + 557056 + 8388608);             // 528*4096*2   = 4325376
    float*    sel = (float*)   (ws + 557056 + 8388608 + 4325376);   // 1024*528*4   = 2162688

    twiddle_kernel        <<<256,  256, 0, stream>>>(Bt2);
    softmax_kernel        <<<NROW, 256, 0, stream>>>(selections, wH);
    transpose_items_kernel<<<NPAD, 256, 0, stream>>>(items, itT);
    sel_gemm_kernel       <<<66,   256, 0, stream>>>(wH, itT, sel);

    const size_t ldsBytes = (size_t)(NFRM + 1) * ASTRIDE * sizeof(_Float16); // 142416
    resonance_kernel<<<NROW, 256, ldsBytes, stream>>>(sel, Bt2, out);
}
